// GATClassifier_25348896981471
// MI455X (gfx1250) — matrix operand order repaired and checked by reading, not compiled, not run on hardware
//
#include <hip/hip_runtime.h>
#include <hip/hip_bf16.h>

// ---------------- problem constants (fixed by reference) ----------------
constexpr int N_NODES = 1024;
constexpr int E_EDGES = 16384;
constexpr int IN_DIM  = 1300;
constexpr int IN_PAD  = 1312;         // IN_DIM padded to multiple of 32
constexpr int HEADS   = 16;
constexpr int HDIM    = 512;          // per-head dim
constexpr int GROUPS  = 8;
constexpr int FDIM    = HEADS * HDIM; // 8192
constexpr int ETOT    = E_EDGES + N_NODES; // self loops appended
constexpr float SLOPE = 0.2f;

typedef __attribute__((ext_vector_type(16))) _Float16 v16h;
typedef __attribute__((ext_vector_type(8)))  _Float16 v8h;
typedef __attribute__((ext_vector_type(8)))  float    v8f;

// ---------------- monotone float<->u32 map for atomic max ----------------
__device__ __forceinline__ unsigned fkey(float f) {
  unsigned u = __float_as_uint(f);
  return (u & 0x80000000u) ? ~u : (u | 0x80000000u);
}
__device__ __forceinline__ float funkey(unsigned u) {
  return (u & 0x80000000u) ? __uint_as_float(u & 0x7FFFFFFFu)
                           : __uint_as_float(~u);
}

// ---------------- f32 -> f16 row-major convert with K padding ----------------
// out[m][k] for k < Kp:  (k < K) ? (f16)in[m][k] : 0
__global__ __launch_bounds__(256)
void convert_pad_f16(const float* __restrict__ in, _Float16* __restrict__ out,
                     int K, int Kp) {
  int m = blockIdx.y;
  int k = blockIdx.x * 256 + threadIdx.x;
  if (k < Kp)
    out[(size_t)m * Kp + k] = (k < K) ? (_Float16)in[(size_t)m * K + k]
                                      : (_Float16)0.0f;
}

// ---------------- f32 [K][Nc] -> f16 transposed [Nc][Kp] (zero pad K tail) ----
__global__ __launch_bounds__(256)
void transpose_f16(const float* __restrict__ in, _Float16* __restrict__ out,
                   int K, int Nc, int Kp) {
  __shared__ float t[32][33];
  int kBase = blockIdx.y * 32;
  int nBase = blockIdx.x * 32;
  int tx = threadIdx.x, ty = threadIdx.y;   // block (32, 8)
  #pragma unroll
  for (int j = 0; j < 4; ++j) {
    int k = kBase + ty + j * 8;
    t[ty + j * 8][tx] = (k < K) ? in[(size_t)k * Nc + nBase + tx] : 0.0f;
  }
  __syncthreads();
  #pragma unroll
  for (int j = 0; j < 4; ++j) {
    int n = nBase + ty + j * 8;
    out[(size_t)n * Kp + kBase + tx] = (_Float16)t[tx][ty + j * 8];
  }
}

// ---------------- WMMA GEMM: C[M,Nc] = A[M,Kp] * Bt[Nc,Kp]^T ----------------
// A row-major f16 [M][Kp]; Bt transposed f16 [Nc][Kp]; C f32 [M][Nc].
// Kp multiple of 32 (pre-padded) -> no bounds checks, fully vector staging.
// Block: 256 threads = 8 waves (4x2). Tile 128x64, K-step 32.
constexpr int TM = 128, TN = 64, TK = 32;

__global__ __launch_bounds__(256)
void gat_wmma_gemm(const _Float16* __restrict__ A, const _Float16* __restrict__ Bt,
                   float* __restrict__ C, int Nc, int Kp) {
  __shared__ __align__(32) _Float16 As[TM][TK]; // [m][k]
  __shared__ __align__(32) _Float16 Bs[TN][TK]; // [n][k]

  const int tid  = threadIdx.x;
  const int lane = tid & 31;
  const int wave = tid >> 5;
  const int wm   = wave >> 1;      // 0..3
  const int wn   = wave & 1;       // 0..1
  const int lrow = lane & 15;
  const int hi   = lane >> 4;

  const int tileM = blockIdx.y * TM;
  const int tileN = blockIdx.x * TN;

  v8f acc[2][2] = {};

  for (int k0 = 0; k0 < Kp; k0 += TK) {
    // stage A: 128x32 halves = 2 x (256 threads * v8h)
    #pragma unroll
    for (int rep = 0; rep < 2; ++rep) {
      int base = rep * 2048 + tid * 8;
      int m = base >> 5, k = base & 31;           // k in {0,8,16,24}
      *(v8h*)&As[m][k] =
          *(const v8h*)&A[(size_t)(tileM + m) * Kp + k0 + k];
    }
    // stage B: 64x32 halves = 1 x (256 threads * v8h)
    {
      int base = tid * 8;
      int n = base >> 5, k = base & 31;
      *(v8h*)&Bs[n][k] =
          *(const v8h*)&Bt[(size_t)(tileN + n) * Kp + k0 + k];
    }
    if (k0 + TK < Kp) {
      __builtin_prefetch(&A[(size_t)(tileM + (tid & 127)) * Kp + (k0 + TK)], 0, 1);
      __builtin_prefetch(&Bt[(size_t)(tileN + (tid & 63)) * Kp + (k0 + TK)], 0, 1);
    }
    __syncthreads();

    #pragma unroll
    for (int fm = 0; fm < 2; ++fm) {
      const int m = wm * 32 + fm * 16 + lrow;
      // A frag: lane = M; lanes 0-15 K=0..15, lanes 16-31 K=16..31
      v16h a = *(const v16h*)(&As[m][hi * 16]);

      #pragma unroll
      for (int fn = 0; fn < 2; ++fn) {
        const int n = wn * 32 + fn * 16 + lrow;
        // B frag: lane = N; lanes 0-15 K=0..15, lanes 16-31 K=16..31
        v16h b = *(const v16h*)(&Bs[n][hi * 16]);
        acc[fm][fn] = __builtin_amdgcn_wmma_f32_16x16x32_f16(
            false, a, false, b, (short)0, acc[fm][fn], false, false);
      }
    }
    __syncthreads();
  }

  // C layout: VGPR r -> M = r + 8*(lane>=16); N = lane&15
  #pragma unroll
  for (int fm = 0; fm < 2; ++fm)
    #pragma unroll
    for (int fn = 0; fn < 2; ++fn)
      #pragma unroll
      for (int r = 0; r < 8; ++r) {
        int row = tileM + wm * 32 + fm * 16 + r + hi * 8;
        int col = tileN + wn * 32 + fn * 16 + lrow;
        C[(size_t)row * Nc + col] = acc[fm][fn][r];
      }
}

// ---------------- attention scores ----------------
__global__ __launch_bounds__(256)
void gat_scores(const float* __restrict__ feat, const float* __restrict__ asrc,
                const float* __restrict__ adst, float* __restrict__ ssrc,
                float* __restrict__ sdst) {
  int gwave = (blockIdx.x * blockDim.x + threadIdx.x) >> 5;
  int lane  = threadIdx.x & 31;
  if (gwave >= N_NODES * HEADS) return;
  int n = gwave >> 4;
  int h = gwave & (HEADS - 1);
  const float* fp = feat + (size_t)n * FDIM + h * HDIM;
  const float* as = asrc + h * HDIM;
  const float* ad = adst + h * HDIM;
  float s1 = 0.f, s2 = 0.f;
  for (int d = lane; d < HDIM; d += 32) {
    float v = fp[d];
    s1 += v * as[d];
    s2 += v * ad[d];
  }
  #pragma unroll
  for (int off = 16; off; off >>= 1) {
    s1 += __shfl_xor(s1, off, 32);
    s2 += __shfl_xor(s2, off, 32);
  }
  if (lane == 0) { ssrc[n * HEADS + h] = s1; sdst[n * HEADS + h] = s2; }
}

// ---------------- fills ----------------
__global__ void fill_f32(float* p, float v, int n) {
  int i = blockIdx.x * 256 + threadIdx.x;
  if (i < n) p[i] = v;
}
__global__ void fill_u32(unsigned* p, unsigned v, int n) {
  int i = blockIdx.x * 256 + threadIdx.x;
  if (i < n) p[i] = v;
}

// ---------------- CSR build over destination nodes ----------------
__device__ __forceinline__ int edge_src(const int* ei, int e) {
  return (e < E_EDGES) ? ei[e] : (e - E_EDGES);
}
__device__ __forceinline__ int edge_dst(const int* ei, int e) {
  return (e < E_EDGES) ? ei[E_EDGES + e] : (e - E_EDGES);
}

__global__ __launch_bounds__(256)
void csr_count(const int* __restrict__ ei, int* __restrict__ deg) {
  int e = blockIdx.x * 256 + threadIdx.x;
  if (e < ETOT) atomicAdd(&deg[edge_dst(ei, e)], 1);
}
__global__ void csr_scan(const int* __restrict__ deg, int* __restrict__ off) {
  if (blockIdx.x == 0 && threadIdx.x == 0) {
    int a = 0;
    for (int i = 0; i < N_NODES; ++i) { off[i] = a; a += deg[i]; }
    off[N_NODES] = a;
  }
}
__global__ __launch_bounds__(256)
void csr_fill(const int* __restrict__ ei, const int* __restrict__ off,
              int* __restrict__ cursor, int* __restrict__ slotE,
              int* __restrict__ slotS) {
  int e = blockIdx.x * 256 + threadIdx.x;
  if (e >= ETOT) return;
  int d = edge_dst(ei, e);
  int pos = atomicAdd(&cursor[d], 1);
  int slot = off[d] + pos;
  slotE[slot] = e;
  slotS[slot] = edge_src(ei, e);
}

// ---------------- edge pass 1: segment max of leaky-relu score ----------------
__global__ __launch_bounds__(256)
void gat_edge_max(const float* __restrict__ ssrc, const float* __restrict__ sdst,
                  const int* __restrict__ ei, unsigned* __restrict__ umax) {
  int idx = blockIdx.x * 256 + threadIdx.x;
  if (idx >= ETOT * HEADS) return;
  int e = idx >> 4, h = idx & (HEADS - 1);
  int s = edge_src(ei, e), d = edge_dst(ei, e);
  float v = ssrc[s * HEADS + h] + sdst[d * HEADS + h];
  v = (v > 0.f) ? v : SLOPE * v;
  atomicMax(&umax[d * HEADS + h], fkey(v));
}

// ---------------- edge pass 2: p = exp(e - m[dst]); denom[dst] += p ----------
__global__ __launch_bounds__(256)
void gat_edge_exp(const float* __restrict__ ssrc, const float* __restrict__ sdst,
                  const int* __restrict__ ei, const unsigned* __restrict__ umax,
                  float* __restrict__ pexp, float* __restrict__ denom) {
  int idx = blockIdx.x * 256 + threadIdx.x;
  if (idx >= ETOT * HEADS) return;
  int e = idx >> 4, h = idx & (HEADS - 1);
  int s = edge_src(ei, e), d = edge_dst(ei, e);
  float v = ssrc[s * HEADS + h] + sdst[d * HEADS + h];
  v = (v > 0.f) ? v : SLOPE * v;
  float p = __expf(v - funkey(umax[d * HEADS + h]));
  pexp[e * HEADS + h] = p;
  atomicAdd(&denom[d * HEADS + h], p);
}

// ---------------- dst-parallel aggregation (no atomics) ----------------
// grid (N_NODES, FDIM/1024); each thread owns 4 consecutive features.
__global__ __launch_bounds__(256)
void gat_aggregate(const float* __restrict__ feat, const float* __restrict__ pexp,
                   const float* __restrict__ denom, const int* __restrict__ off,
                   const int* __restrict__ slotE, const int* __restrict__ slotS,
                   float* __restrict__ outacc) {
  int d = blockIdx.x;
  int tid = threadIdx.x;
  int fbase = blockIdx.y * 1024 + tid * 4;
  int h = fbase >> 9;              // head of all 4 features (512 | 4-aligned)

  __shared__ float invDen[HEADS];
  __shared__ float alphaS[HEADS];
  if (tid < HEADS) invDen[tid] = 1.0f / denom[d * HEADS + tid];

  float4 acc = make_float4(0.f, 0.f, 0.f, 0.f);
  int beg = off[d], end = off[d + 1];
  for (int j = beg; j < end; ++j) {
    __syncthreads();               // protect alphaS against previous readers
    if (tid < HEADS)
      alphaS[tid] = pexp[slotE[j] * HEADS + tid] * invDen[tid];
    __syncthreads();
    const float4 v = *(const float4*)&feat[(size_t)slotS[j] * FDIM + fbase];
    float a = alphaS[h];
    acc.x += a * v.x; acc.y += a * v.y; acc.z += a * v.z; acc.w += a * v.w;
  }
  *(float4*)&outacc[(size_t)d * FDIM + fbase] = acc;
}

// ---------------- bias + relu (in place) ----------------
__global__ __launch_bounds__(256)
void gat_bias_relu(float* __restrict__ h, const float* __restrict__ b) {
  int i = blockIdx.x * 256 + threadIdx.x;
  if (i < N_NODES * FDIM) {
    float v = h[i] + b[i & (FDIM - 1)];
    h[i] = v > 0.f ? v : 0.f;
  }
}

// ---------------- mean pool ----------------
__global__ __launch_bounds__(256)
void gat_pool(const float* __restrict__ h, const int* __restrict__ batch,
              float* __restrict__ pooled, float* __restrict__ counts) {
  int n = blockIdx.x;
  int g = batch[n];
  if (threadIdx.x == 0) atomicAdd(&counts[g], 1.0f);
  const float* src = h + (size_t)n * FDIM;
  float* dst = pooled + (size_t)g * FDIM;
  for (int f = threadIdx.x; f < FDIM; f += 256)
    atomicAdd(&dst[f], src[f]);
}

// ---------------- final linear ----------------
__global__ __launch_bounds__(256)
void gat_final(const float* __restrict__ pooled, const float* __restrict__ counts,
               const float* __restrict__ lw, const float* __restrict__ lb,
               float* __restrict__ out) {
  int g = blockIdx.x;
  __shared__ float red[256];
  float s = 0.f;
  for (int f = threadIdx.x; f < FDIM; f += 256)
    s += pooled[(size_t)g * FDIM + f] * lw[f];
  red[threadIdx.x] = s;
  __syncthreads();
  for (int st = 128; st; st >>= 1) {
    if (threadIdx.x < st) red[threadIdx.x] += red[threadIdx.x + st];
    __syncthreads();
  }
  if (threadIdx.x == 0) {
    float c = counts[g];
    if (c < 1.f) c = 1.f;
    out[g] = red[0] / c + lb[0];
  }
}

// ---------------- orchestration ----------------
extern "C" void kernel_launch(void* const* d_in, const int* in_sizes, int n_in,
                              void* d_out, int out_size, void* d_ws, size_t ws_size,
                              hipStream_t stream) {
  const float* x    = (const float*)d_in[0];
  const int*   ei   = (const int*)d_in[1];
  const int*   bat  = (const int*)d_in[2];
  const float* W1   = (const float*)d_in[3];
  const float* as1  = (const float*)d_in[4];
  const float* ad1  = (const float*)d_in[5];
  const float* b1   = (const float*)d_in[6];
  const float* W2   = (const float*)d_in[7];
  const float* as2  = (const float*)d_in[8];
  const float* ad2  = (const float*)d_in[9];
  const float* b2   = (const float*)d_in[10];
  const float* lw   = (const float*)d_in[11];
  const float* lb   = (const float*)d_in[12];
  float* out = (float*)d_out;

  // ---- workspace bump allocator (256B aligned slices) ----
  char* p = (char*)d_ws;
  auto alloc = [&](size_t bytes) -> char* {
    char* r = p;
    p += (bytes + 255) & ~(size_t)255;
    return r;
  };
  float*    feat   = (float*)alloc(sizeof(float) * N_NODES * FDIM);
  float*    accum  = (float*)alloc(sizeof(float) * N_NODES * FDIM);
  float*    ssrc   = (float*)alloc(sizeof(float) * N_NODES * HEADS);
  float*    sdst   = (float*)alloc(sizeof(float) * N_NODES * HEADS);
  unsigned* umax   = (unsigned*)alloc(sizeof(unsigned) * N_NODES * HEADS);
  float*    denom  = (float*)alloc(sizeof(float) * N_NODES * HEADS);
  float*    pexp   = (float*)alloc(sizeof(float) * ETOT * HEADS);
  float*    pooled = (float*)alloc(sizeof(float) * GROUPS * FDIM);
  float*    counts = (float*)alloc(sizeof(float) * GROUPS);
  int*      deg    = (int*)alloc(sizeof(int) * N_NODES);
  int*      off    = (int*)alloc(sizeof(int) * (N_NODES + 1));
  int*      cursor = (int*)alloc(sizeof(int) * N_NODES);
  int*      slotE  = (int*)alloc(sizeof(int) * ETOT);
  int*      slotS  = (int*)alloc(sizeof(int) * ETOT);
  _Float16* xh     = (_Float16*)alloc(sizeof(_Float16) * N_NODES * IN_PAD);
  _Float16* w1t    = (_Float16*)alloc(sizeof(_Float16) * (size_t)FDIM * IN_PAD);
  _Float16* w2t    = (_Float16*)alloc(sizeof(_Float16) * (size_t)FDIM * FDIM);
  _Float16* hh     = (_Float16*)alloc(sizeof(_Float16) * (size_t)N_NODES * FDIM);

  const dim3 gemmGrid(FDIM / TN, N_NODES / TM);            // (128, 8)
  const dim3 trBlk(32, 8);
  const int  nhBlocks = (N_NODES * HEADS * 32 + 255) / 256;
  const int  ehBlocks = (ETOT * HEADS + 255) / 256;
  const int  eBlocks  = (ETOT + 255) / 256;
  const int  nfBlocks = (N_NODES * FDIM + 255) / 256;
  const int  nhFill   = (N_NODES * HEADS + 255) / 256;
  const unsigned NEG_INF_KEY = 0x007FFFFFu;                // fkey(-inf)

  // ---- CSR over dst (shared by both layers) ----
  fill_u32<<<(N_NODES + 255) / 256, 256, 0, stream>>>((unsigned*)deg, 0u, N_NODES);
  fill_u32<<<(N_NODES + 255) / 256, 256, 0, stream>>>((unsigned*)cursor, 0u, N_NODES);
  csr_count<<<eBlocks, 256, 0, stream>>>(ei, deg);
  csr_scan<<<1, 1, 0, stream>>>(deg, off);
  csr_fill<<<eBlocks, 256, 0, stream>>>(ei, off, cursor, slotE, slotS);

  // ---- one-time per-call f16 conversions / transposes ----
  convert_pad_f16<<<dim3((IN_PAD + 255) / 256, N_NODES), 256, 0, stream>>>(
      x, xh, IN_DIM, IN_PAD);
  transpose_f16<<<dim3(FDIM / 32, IN_PAD / 32), trBlk, 0, stream>>>(
      W1, w1t, IN_DIM, FDIM, IN_PAD);
  transpose_f16<<<dim3(FDIM / 32, FDIM / 32), trBlk, 0, stream>>>(
      W2, w2t, FDIM, FDIM, FDIM);

  // ================= layer 1 =================
  gat_wmma_gemm<<<gemmGrid, 256, 0, stream>>>(xh, w1t, feat, FDIM, IN_PAD);
  gat_scores<<<nhBlocks, 256, 0, stream>>>(feat, as1, ad1, ssrc, sdst);
  fill_u32<<<nhFill, 256, 0, stream>>>(umax, NEG_INF_KEY, N_NODES * HEADS);
  fill_f32<<<nhFill, 256, 0, stream>>>(denom, 0.f, N_NODES * HEADS);
  gat_edge_max<<<ehBlocks, 256, 0, stream>>>(ssrc, sdst, ei, umax);
  gat_edge_exp<<<ehBlocks, 256, 0, stream>>>(ssrc, sdst, ei, umax, pexp, denom);
  gat_aggregate<<<dim3(N_NODES, FDIM / 1024), 256, 0, stream>>>(
      feat, pexp, denom, off, slotE, slotS, accum);
  gat_bias_relu<<<nfBlocks, 256, 0, stream>>>(accum, b1);

  // ================= layer 2 =================
  convert_pad_f16<<<dim3(FDIM / 256, N_NODES), 256, 0, stream>>>(
      accum, hh, FDIM, FDIM);
  gat_wmma_gemm<<<gemmGrid, 256, 0, stream>>>(hh, w2t, feat, FDIM, FDIM);
  gat_scores<<<nhBlocks, 256, 0, stream>>>(feat, as2, ad2, ssrc, sdst);
  fill_u32<<<nhFill, 256, 0, stream>>>(umax, NEG_INF_KEY, N_NODES * HEADS);
  fill_f32<<<nhFill, 256, 0, stream>>>(denom, 0.f, N_NODES * HEADS);
  gat_edge_max<<<ehBlocks, 256, 0, stream>>>(ssrc, sdst, ei, umax);
  gat_edge_exp<<<ehBlocks, 256, 0, stream>>>(ssrc, sdst, ei, umax, pexp, denom);
  gat_aggregate<<<dim3(N_NODES, FDIM / 1024), 256, 0, stream>>>(
      feat, pexp, denom, off, slotE, slotS, accum);
  gat_bias_relu<<<nfBlocks, 256, 0, stream>>>(accum, b2);

  // ================= pool + head =================
  fill_f32<<<(GROUPS * FDIM + 255) / 256, 256, 0, stream>>>(pooled, 0.f,
                                                            GROUPS * FDIM);
  fill_f32<<<1, 256, 0, stream>>>(counts, 0.f, GROUPS);
  gat_pool<<<N_NODES, 256, 0, stream>>>(accum, bat, pooled, counts);
  gat_final<<<GROUPS, 256, 0, stream>>>(pooled, counts, lw, lb, out);
}
